// DynamicRNNLayerWithAttention_3685081940221
// MI455X (gfx1250) — matrix operand order repaired and checked by reading, not compiled, not run on hardware
//
#include <hip/hip_runtime.h>
#include <hip/hip_bf16.h>
#include <math.h>

typedef __bf16 bf16_t;
typedef bf16_t v16bf __attribute__((ext_vector_type(16)));
typedef float  v8f   __attribute__((ext_vector_type(8)));

#define T_STEPS 1024
#define F_DIM   256
#define H_DIM   256
#define M_DIM   256
#define D_DIM   256
#define K_TOT   768          // F + D + H
#define NG      1024         // 4*H
#define BT      16           // batches per workgroup (one WMMA M-tile)

__device__ __forceinline__ float sigmoidf_(float v) { return 1.0f / (1.0f + __expf(-v)); }

// ---------------- one-time: W (768x1024 f32) -> Wt (1024x768 bf16, transposed) ----------------
__global__ void wt_convert_kernel(const float* __restrict__ W, bf16_t* __restrict__ Wt) {
    int i = blockIdx.x * 256 + threadIdx.x;
    if (i >= K_TOT * NG) return;
    int k = i / NG, n = i % NG;
    Wt[(size_t)n * K_TOT + k] = (bf16_t)W[i];
}

// ---------------- one-time: keys[b][m][h] = sum_d memory[b][m][d] * Wmem[d][h] (fp32) ----------
__global__ void keys_kernel(const float* __restrict__ memory, const float* __restrict__ Wmem,
                            float* __restrict__ keys) {
    __shared__ float row[D_DIM];
    int bm = blockIdx.x;          // b*M + m
    int h  = threadIdx.x;
    row[h] = memory[(size_t)bm * D_DIM + h];
    __syncthreads();
    float acc = 0.0f;
#pragma unroll 4
    for (int d = 0; d < D_DIM; ++d) acc += row[d] * Wmem[(size_t)d * H_DIM + h];
    keys[(size_t)bm * H_DIM + h] = acc;
}

// ---------------- persistent recurrent kernel: 4 blocks x 256 threads (8 wave32) ---------------
__global__ __launch_bounds__(256) void rnn_attn_kernel(
    const float* __restrict__ x, const float* __restrict__ memory,
    const float* __restrict__ bias, const int* __restrict__ seq_len,
    const float* __restrict__ keys, const bf16_t* __restrict__ Wt,
    float* __restrict__ out) {
    __shared__ __align__(16) bf16_t cellin[BT][K_TOT];  // 24 KB  [x_t | attn | h] in bf16
    __shared__ __align__(16) float  gates[BT][NG];      // 64 KB  (cols 0..255 reused as h_new,
                                                        //         cols 768..1023 reused as score/align)
    __shared__ float hbuf[BT][H_DIM];                   // 16 KB carried h
    __shared__ float attn[BT][D_DIM];                   // 16 KB carried attention

    const int tid  = threadIdx.x;          // 0..255
    const int bg   = blockIdx.x;           // 0..3
    const int b0   = bg * BT;
    const int wave = tid >> 5;
    const int lane = tid & 31;
    const int lm   = lane & 15;            // tile row (A.M / B.N / D.N)
    const int lhi  = lane >> 4;            // half-wave select

    float creg[BT];                        // LSTM cell state, (b=p, h=tid) owned per-thread
    int   sl[BT];
#pragma unroll
    for (int p = 0; p < BT; ++p) {
        creg[p] = 0.0f;
        hbuf[p][tid] = 0.0f;
        attn[p][tid] = 0.0f;
        sl[p] = seq_len[b0 + p];
    }
    __syncthreads();

#pragma unroll 1
    for (int t = 0; t < T_STEPS; ++t) {
        // -------- phase 1: build cell_in = [x_t | attn | h] as bf16 --------
#pragma unroll 1
        for (int p = 0; p < BT; ++p) {
            float xv = x[((size_t)(b0 + p) * T_STEPS + t) * F_DIM + tid];
            cellin[p][tid]                 = (bf16_t)xv;
            cellin[p][F_DIM + tid]         = (bf16_t)attn[p][tid];
            cellin[p][F_DIM + D_DIM + tid] = (bf16_t)hbuf[p][tid];
            if (t + 1 < T_STEPS)   // stream next step's x from HBM behind this step's compute
                __builtin_prefetch(&x[((size_t)(b0 + p) * T_STEPS + (t + 1)) * F_DIM + tid], 0, 1);
        }
        __syncthreads();

        // -------- phase 2: gates[16x1024] = cellin[16x768] @ W  (WMMA bf16) --------
        // kb outer / 8 N-tiles inner with 8 live accumulators: A frag loaded once per kb,
        // 8 unrolled B-load+WMMA pairs give the scheduler room for partial loadcnt waits.
        {
            v8f acc[8];
            v8f zero = {};
#pragma unroll
            for (int i = 0; i < 8; ++i) acc[i] = zero;
            const bf16_t* arow  = &cellin[lm][lhi * 8];
            const bf16_t* bbase = Wt + (size_t)(wave * 16 + lm) * K_TOT + lhi * 8;
#pragma unroll 2
            for (int kb = 0; kb < K_TOT / 32; ++kb) {
                union { unsigned u[8]; v16bf v; } af;
                const unsigned* a32 = (const unsigned*)(arow + kb * 32);
                // A 16-bit 16x32 layout: dwords at K = kbase+{0,2,4,6} and kbase+{16,18,20,22}
                af.u[0] = a32[0];  af.u[1] = a32[1];  af.u[2] = a32[2];  af.u[3] = a32[3];
                af.u[4] = a32[8];  af.u[5] = a32[9];  af.u[6] = a32[10]; af.u[7] = a32[11];
#pragma unroll
                for (int nti = 0; nti < 8; ++nti) {
                    union { unsigned u[8]; v16bf v; } bf;
                    const unsigned* b32 =
                        (const unsigned*)(bbase + (size_t)nti * 128 * K_TOT + kb * 32);
#pragma unroll
                    for (int v = 0; v < 8; ++v) bf.u[v] = b32[v + (v & 4)];  // B: same K layout as A (8*lhi+{0..7}, 16+8*lhi+{0..7})
                    acc[nti] = __builtin_amdgcn_wmma_f32_16x16x32_bf16(
                        false, af.v, false, bf.v, (short)0, acc[nti], false, false);
                }
            }
            // D layout: element (M = v + 8*lhi, N = lm)
#pragma unroll
            for (int nti = 0; nti < 8; ++nti) {
                int nt = wave + 8 * nti;
#pragma unroll
                for (int v = 0; v < 8; ++v) gates[v + 8 * lhi][nt * 16 + lm] = acc[nti][v];
            }
        }
        __syncthreads();

        // -------- phase 3: LSTM elementwise (b=p, h=tid); h_new stored into gates i-slot ------
#pragma unroll 1
        for (int p = 0; p < BT; ++p) {
            float gi = gates[p][tid]           + bias[tid];
            float gj = gates[p][H_DIM + tid]   + bias[H_DIM + tid];
            float gf = gates[p][2*H_DIM + tid] + bias[2*H_DIM + tid];
            float go = gates[p][3*H_DIM + tid] + bias[3*H_DIM + tid];
            float cn = creg[p] * sigmoidf_(gf + 1.0f) + sigmoidf_(gi) * tanhf(gj);
            float hn = tanhf(cn) * sigmoidf_(go);
            bool valid = (t < sl[p]);
            if (valid) { creg[p] = cn; hbuf[p][tid] = hn; }
            gates[p][tid] = hn;   // h_new buffer (aliases consumed i-slot; (p,tid) private)
        }
        __syncthreads();

        // -------- phase 4: score[p][m=tid] = h_new[p] . keys[b0+p][m]  (b128 fp32 loads) ------
#pragma unroll 1
        for (int p = 0; p < BT; ++p) {
            const float4* krow = (const float4*)(keys + ((size_t)(b0 + p) * M_DIM + tid) * H_DIM);
            const float4* hn4  = (const float4*)(&gates[p][0]);
            float acc = 0.0f;
#pragma unroll 8
            for (int q = 0; q < H_DIM / 4; ++q) {
                float4 k4 = krow[q];
                float4 h4 = hn4[q];
                acc += h4.x * k4.x + h4.y * k4.y + h4.z * k4.z + h4.w * k4.w;
            }
            gates[p][3*H_DIM + tid] = acc;    // score aliases consumed o-slot
        }
        __syncthreads();

        // -------- phase 5: softmax over m per row; 8 waves handle 16 rows -----------
        for (int r = wave; r < BT; r += 8) {
            float* srow = &gates[r][3*H_DIM];
            float mx = -INFINITY;
            for (int m = lane; m < M_DIM; m += 32) mx = fmaxf(mx, srow[m]);
#pragma unroll
            for (int off = 16; off > 0; off >>= 1) mx = fmaxf(mx, __shfl_xor(mx, off, 32));
            float sum = 0.0f;
            for (int m = lane; m < M_DIM; m += 32) {
                float e = __expf(srow[m] - mx);
                srow[m] = e;
                sum += e;
            }
#pragma unroll
            for (int off = 16; off > 0; off >>= 1) sum += __shfl_xor(sum, off, 32);
            float inv = 1.0f / sum;
            for (int m = lane; m < M_DIM; m += 32) srow[m] *= inv;
        }
        __syncthreads();

        // -------- phase 6: ctx[p][d=tid] = align[p] . memory[b0+p][:,d]; masked attn update ---
#pragma unroll 1
        for (int p = 0; p < BT; ++p) {
            const float* al   = &gates[p][3*H_DIM];
            const float* mcol = memory + (size_t)(b0 + p) * M_DIM * D_DIM + tid;
            float a0 = 0.f, a1 = 0.f, a2 = 0.f, a3 = 0.f;
#pragma unroll 2
            for (int m = 0; m < M_DIM; m += 4) {
                a0 += al[m]     * mcol[(size_t)m       * D_DIM];
                a1 += al[m + 1] * mcol[(size_t)(m + 1) * D_DIM];
                a2 += al[m + 2] * mcol[(size_t)(m + 2) * D_DIM];
                a3 += al[m + 3] * mcol[(size_t)(m + 3) * D_DIM];
            }
            float acc = (a0 + a1) + (a2 + a3);
            if (t < sl[p]) attn[p][tid] = acc;
        }
        __syncthreads();
    }

    // -------- write final attention state: out[64][256] --------
#pragma unroll
    for (int p = 0; p < BT; ++p) out[(size_t)(b0 + p) * D_DIM + tid] = attn[p][tid];
}

// ----------------------------------------------------------------------------------------------
extern "C" void kernel_launch(void* const* d_in, const int* in_sizes, int n_in,
                              void* d_out, int out_size, void* d_ws, size_t ws_size,
                              hipStream_t stream) {
    const float* x      = (const float*)d_in[0];   // [64,1024,256]
    const float* memory = (const float*)d_in[1];   // [64,256,256]
    const float* W      = (const float*)d_in[2];   // [768,1024]
    const float* bias   = (const float*)d_in[3];   // [1024]
    const float* Wmem   = (const float*)d_in[4];   // [256,256]
    const int*   seqlen = (const int*)d_in[5];     // [64]
    float* out = (float*)d_out;                    // [64,256]

    bf16_t* Wt   = (bf16_t*)d_ws;                                               // 1.5 MB
    float*  keys = (float*)((char*)d_ws + (size_t)K_TOT * NG * sizeof(bf16_t)); // 16 MB

    wt_convert_kernel<<<(K_TOT * NG + 255) / 256, 256, 0, stream>>>(W, Wt);
    keys_kernel<<<64 * M_DIM, 256, 0, stream>>>(memory, Wmem, keys);
    rnn_attn_kernel<<<4, 256, 0, stream>>>(x, memory, bias, seqlen, keys, Wt, out);

    (void)in_sizes; (void)n_in; (void)out_size; (void)ws_size;
}
